// CrossFrameAttentionProcessor2_0_28312424415681
// MI455X (gfx1250) — compile-verified, matrix operand order checked by reading, not run on hardware
//
#include <hip/hip_runtime.h>
#include <hip/hip_bf16.h>
#include <math.h>

typedef __attribute__((ext_vector_type(16))) _Float16 v16h;
typedef __attribute__((ext_vector_type(8)))  float    v8f;

#define NB    5          // frames (B)
#define SEQ   1024
#define CH    320
#define NH    8
#define HD    40
#define DPAD  64
#define KF    3
#define LKEYS (KF * SEQ)
#define ROWS  (NB * SEQ)  // 5120

// ---------------------------------------------------------------------------
// Fragment helpers (CDNA5 ISA 7.12.2 layouts, wave32)
// A-frag 16x32 f16: lane holds row = lane%16; VGPR j holds K-pair
//   K = 16*(j>=4) + 2*(j%4) + 8*(lane/16) .. +1
// Same mapping used for B-frags by pre-transposing B so "rows" are N-columns.
// ---------------------------------------------------------------------------
__device__ __forceinline__ v16h load_frag16(const _Float16* __restrict__ base,
                                            int ld, int lane) {
  const int hi = (lane >> 4) & 1;
  const int r  = lane & 15;
  const _Float16* row = base + (size_t)r * (size_t)ld + (hi << 3);
  v16h a;
#pragma unroll
  for (int j = 0; j < 8; ++j) {
    const int k = ((j >> 2) << 4) + ((j & 3) << 1);
    a[2 * j]     = row[k];
    a[2 * j + 1] = row[k + 1];
  }
  return a;
}

__device__ __forceinline__ v8f wmma_f16(v16h a, v16h b, v8f c) {
  return __builtin_amdgcn_wmma_f32_16x16x32_f16(false, a, false, b,
                                                (short)0, c, false, false);
}

// ---------------------------------------------------------------------------
// gfx1250 async global->LDS staging (ASYNCcnt path)
// One instruction moves 32 lanes x 16B = 512B; 8 instructions = 4KB chunk.
// ---------------------------------------------------------------------------
__device__ __forceinline__ void async_copy_4k(unsigned lds_byte_off,
                                              const _Float16* g, int lane) {
  unsigned loff = lds_byte_off + (unsigned)lane * 16u;
  unsigned long long gp = (unsigned long long)g + (unsigned long long)lane * 16ull;
#pragma unroll
  for (int t = 0; t < 8; ++t) {
    asm volatile("global_load_async_to_lds_b128 %0, %1, off"
                 :: "v"(loff), "v"(gp) : "memory");
    loff += 512u;
    gp   += 512ull;
  }
}
__device__ __forceinline__ void wait_async_le8() {
  asm volatile("s_wait_asynccnt 0x8" ::: "memory");
}
__device__ __forceinline__ void wait_async_le0() {
  asm volatile("s_wait_asynccnt 0x0" ::: "memory");
}

// ---------------------------------------------------------------------------
// Prep kernels
// ---------------------------------------------------------------------------
__global__ void k_f32_to_f16(const float* __restrict__ src,
                             _Float16* __restrict__ dst, int n) {
  int i = blockIdx.x * blockDim.x + threadIdx.x;
  if (i < n) dst[i] = (_Float16)src[i];
}

__global__ void k_transpose_w(const float* __restrict__ w0, const float* __restrict__ w1,
                              const float* __restrict__ w2, const float* __restrict__ w3,
                              _Float16* __restrict__ t0, _Float16* __restrict__ t1,
                              _Float16* __restrict__ t2, _Float16* __restrict__ t3) {
  int i = blockIdx.x * blockDim.x + threadIdx.x;
  if (i >= 4 * CH * CH) return;
  const int w = i / (CH * CH);
  const int r = i - w * (CH * CH);
  const int k = r / CH;
  const int n = r - k * CH;
  const float* ws = (w == 0) ? w0 : (w == 1) ? w1 : (w == 2) ? w2 : w3;
  _Float16*    ts = (w == 0) ? t0 : (w == 1) ? t1 : (w == 2) ? t2 : t3;
  ts[(size_t)n * CH + k] = (_Float16)ws[(size_t)k * CH + n];
}

__global__ void k_zero16(uint4* __restrict__ p, int n16) {
  int i = blockIdx.x * blockDim.x + threadIdx.x;
  if (i < n16) p[i] = make_uint4(0u, 0u, 0u, 0u);
}

// ---------------------------------------------------------------------------
// QKV projection: 32x32 output per wave (2x2 WMMA tiles, 4 WMMAs / K-step).
//   Q,K : (b,h,s,DPAD)   V : (b,h,DPAD,s)  (transposed for PV B-frags)
// Q is pre-scaled by 1/sqrt(d) so the attention kernel skips score scaling.
// ---------------------------------------------------------------------------
__global__ __launch_bounds__(256) void k_qkv_gemm(
    const _Float16* __restrict__ hs,
    const _Float16* __restrict__ wtq, const _Float16* __restrict__ wtk,
    const _Float16* __restrict__ wtv,
    _Float16* __restrict__ Qp, _Float16* __restrict__ Kp,
    _Float16* __restrict__ Vt) {
  const int lane = threadIdx.x & 31;
  const int gid  = blockIdx.x * 8 + (threadIdx.x >> 5);
  const int which = gid / 1600;               // 160 mblocks * 10 nblocks
  int rem = gid - which * 1600;
  const int mb = rem / 10;
  const int nb = rem - mb * 10;
  const int m0 = mb * 32, n0 = nb * 32;
  const _Float16* wt = (which == 0) ? wtq : (which == 1) ? wtk : wtv;

  v8f a00 = {0.f,0.f,0.f,0.f,0.f,0.f,0.f,0.f};
  v8f a01 = a00, a10 = a00, a11 = a00;
#pragma unroll 2
  for (int k0 = 0; k0 < CH; k0 += 32) {
    v16h fa0 = load_frag16(hs + (size_t)m0 * CH + k0, CH, lane);
    v16h fa1 = load_frag16(hs + (size_t)(m0 + 16) * CH + k0, CH, lane);
    v16h fb0 = load_frag16(wt + (size_t)n0 * CH + k0, CH, lane);
    v16h fb1 = load_frag16(wt + (size_t)(n0 + 16) * CH + k0, CH, lane);
    a00 = wmma_f16(fa0, fb0, a00);
    a01 = wmma_f16(fa0, fb1, a01);
    a10 = wmma_f16(fa1, fb0, a10);
    a11 = wmma_f16(fa1, fb1, a11);
  }

  const float osc = (which == 0) ? 0.15811388300841897f : 1.0f;  // 1/sqrt(40)
  const int col = lane & 15, hi = (lane >> 4) & 1;
  const int bb = m0 >> 10;  // 32 | 1024 -> whole 32-row block in one frame
#pragma unroll
  for (int ti = 0; ti < 2; ++ti) {
    const int sb = ((m0 + ti * 16) & 1023) + 8 * hi;
#pragma unroll
    for (int tj = 0; tj < 2; ++tj) {
      const v8f acc = (ti == 0) ? ((tj == 0) ? a00 : a01)
                                : ((tj == 0) ? a10 : a11);
      const int c  = n0 + tj * 16 + col;
      const int h  = c / HD;
      const int dd = c - h * HD;
#pragma unroll
      for (int r = 0; r < 8; ++r) {
        const int s = sb + r;
        const _Float16 vv = (_Float16)(acc[r] * osc);
        if (which == 0)
          Qp[(((size_t)bb * NH + h) * SEQ + s) * DPAD + dd] = vv;
        else if (which == 1)
          Kp[(((size_t)bb * NH + h) * SEQ + s) * DPAD + dd] = vv;
        else
          Vt[(((size_t)bb * NH + h) * DPAD + dd) * SEQ + s] = vv;
      }
    }
  }
}

// ---------------------------------------------------------------------------
// Flash attention: one wave per (b, h, 16-row q tile); online softmax;
// K chunks double-buffered in LDS via global_load_async_to_lds_b128.
// ---------------------------------------------------------------------------
__global__ __launch_bounds__(32) void k_attn(
    const _Float16* __restrict__ Qp, const _Float16* __restrict__ Kp,
    const _Float16* __restrict__ Vt, const long long* __restrict__ ids,
    _Float16* __restrict__ attn) {
  // Single shared object so static LDS layout starts at byte offset 0:
  //   [0      .. 8191 ]  kbuf[2][32][64] f16 (double-buffered K chunks)
  //   [8192   .. 9215 ]  p_lds[16][32]   f16 (P staging, D->A re-swizzle)
  __shared__ _Float16 smem[2 * 32 * DPAD + 16 * 32];
  _Float16* kbuf0 = smem;
  _Float16* pbuf  = smem + 2 * 32 * DPAD;

  const int lane = threadIdx.x & 31;
  const int col = lane & 15, hi = (lane >> 4) & 1;
  int bid = blockIdx.x;
  const int qt = bid & 63; bid >>= 6;
  const int h  = bid & 7;  bid >>= 3;
  const int b  = bid;
  const int q0 = qt * 16;

  const long long f0 = ids[b * KF + 0];
  const long long f1 = ids[b * KF + 1];
  const long long f2 = ids[b * KF + 2];

  const _Float16* qbase = Qp + (((size_t)b * NH + h) * SEQ + q0) * DPAD;
  const v16h aq0 = load_frag16(qbase, DPAD, lane);
  const v16h aq1 = load_frag16(qbase + 32, DPAD, lane);

  v8f o0 = {0.f,0.f,0.f,0.f,0.f,0.f,0.f,0.f};
  v8f o1 = o0, o2 = o0;
  float m[8], l[8];
#pragma unroll
  for (int r = 0; r < 8; ++r) { m[r] = -3.0e38f; l[r] = 0.f; }

  // issue async stage of first K chunk (frame f0, keys 0..31)
  {
    const _Float16* kb0 = Kp + (((size_t)f0 * NH + h) * SEQ) * DPAD;
    async_copy_4k(0u, kb0, lane);
  }

  int cur = 0;
  for (int kc = 0; kc < LKEYS; kc += 32) {
    const int fi = kc >> 10;
    const long long f = (fi == 0) ? f0 : (fi == 1) ? f1 : f2;
    const int sk = kc & 1023;

    // stage next chunk while this one is consumed
    if (kc + 32 < LKEYS) {
      const int kn = kc + 32;
      const int fin = kn >> 10;
      const long long fnx = (fin == 0) ? f0 : (fin == 1) ? f1 : f2;
      const _Float16* kbn =
          Kp + (((size_t)fnx * NH + h) * SEQ + (kn & 1023)) * DPAD;
      async_copy_4k((unsigned)((cur ^ 1) * 4096), kbn, lane);
      wait_async_le8();   // first 8 (current chunk) complete, in-order
    } else {
      wait_async_le0();
    }

    // prefetch next V chunk rows into L2/L0
    if (kc + 32 < LKEYS) {
      const int kn = kc + 32;
      const int fin = kn >> 10;
      const long long fnx = (fin == 0) ? f0 : (fin == 1) ? f1 : f2;
      const _Float16* vpn =
          Vt + (((size_t)fnx * NH + h) * DPAD) * SEQ + (kn & 1023);
      __builtin_prefetch((const void*)vpn, 0, 1);
      __builtin_prefetch((const void*)(vpn + (size_t)16 * SEQ), 0, 1);
      __builtin_prefetch((const void*)(vpn + (size_t)32 * SEQ), 0, 1);
    }

    const _Float16* kb = kbuf0 + cur * (32 * DPAD);
    const v16h bk00 = load_frag16(kb, DPAD, lane);
    const v16h bk01 = load_frag16(kb + 32, DPAD, lane);
    const v16h bk10 = load_frag16(kb + 16 * DPAD, DPAD, lane);
    const v16h bk11 = load_frag16(kb + 16 * DPAD + 32, DPAD, lane);

    v8f s0 = {0.f,0.f,0.f,0.f,0.f,0.f,0.f,0.f};
    v8f s1 = s0;
    s0 = wmma_f16(aq0, bk00, s0);
    s0 = wmma_f16(aq1, bk01, s0);
    s1 = wmma_f16(aq0, bk10, s1);
    s1 = wmma_f16(aq1, bk11, s1);

#pragma unroll
    for (int r = 0; r < 8; ++r) {
      const float a  = s0[r];   // Q pre-scaled by 1/sqrt(d)
      const float bb = s1[r];
      float mx = fmaxf(a, bb);
#pragma unroll
      for (int off = 1; off < 16; off <<= 1) mx = fmaxf(mx, __shfl_xor(mx, off, 16));
      const float mnew = fmaxf(m[r], mx);
      const float corr = __expf(m[r] - mnew);
      m[r] = mnew;
      const float p0 = __expf(a - mnew);
      const float p1 = __expf(bb - mnew);
      float rs = p0 + p1;
#pragma unroll
      for (int off = 1; off < 16; off <<= 1) rs += __shfl_xor(rs, off, 16);
      l[r] = l[r] * corr + rs;
      o0[r] *= corr; o1[r] *= corr; o2[r] *= corr;
      const int row = r + 8 * hi;
      pbuf[row * 32 + col]      = (_Float16)p0;
      pbuf[row * 32 + 16 + col] = (_Float16)p1;
    }
    __syncthreads();

    const v16h ap = load_frag16(pbuf, 32, lane);
    const _Float16* vb = Vt + (((size_t)f * NH + h) * DPAD) * SEQ + sk;
    const v16h bv0 = load_frag16(vb, SEQ, lane);
    const v16h bv1 = load_frag16(vb + (size_t)16 * SEQ, SEQ, lane);
    const v16h bv2 = load_frag16(vb + (size_t)32 * SEQ, SEQ, lane);
    o0 = wmma_f16(ap, bv0, o0);
    o1 = wmma_f16(ap, bv1, o1);
    o2 = wmma_f16(ap, bv2, o2);
    __syncthreads();

    cur ^= 1;
  }

#pragma unroll
  for (int r = 0; r < 8; ++r) {
    const float invl = (l[r] > 0.f) ? 1.f / l[r] : 0.f;
    const int s = q0 + r + 8 * hi;
    _Float16* dst = attn + ((size_t)b * SEQ + s) * CH + h * HD;
    dst[col]      = (_Float16)(o0[r] * invl);
    dst[16 + col] = (_Float16)(o1[r] * invl);
    if (32 + col < HD) dst[32 + col] = (_Float16)(o2[r] * invl);
  }
}

// ---------------------------------------------------------------------------
// Output projection + bias + residual (fp32 out), 32x32 per wave
// ---------------------------------------------------------------------------
__global__ __launch_bounds__(256) void k_out_gemm(
    const _Float16* __restrict__ attn, const _Float16* __restrict__ wto,
    const float* __restrict__ bo, const float* __restrict__ hs,
    float* __restrict__ out) {
  const int lane = threadIdx.x & 31;
  const int gid  = blockIdx.x * 8 + (threadIdx.x >> 5);
  const int mb = gid / 10;
  const int nb = gid - mb * 10;
  const int m0 = mb * 32, n0 = nb * 32;

  v8f a00 = {0.f,0.f,0.f,0.f,0.f,0.f,0.f,0.f};
  v8f a01 = a00, a10 = a00, a11 = a00;
#pragma unroll 2
  for (int k0 = 0; k0 < CH; k0 += 32) {
    v16h fa0 = load_frag16(attn + (size_t)m0 * CH + k0, CH, lane);
    v16h fa1 = load_frag16(attn + (size_t)(m0 + 16) * CH + k0, CH, lane);
    v16h fb0 = load_frag16(wto + (size_t)n0 * CH + k0, CH, lane);
    v16h fb1 = load_frag16(wto + (size_t)(n0 + 16) * CH + k0, CH, lane);
    a00 = wmma_f16(fa0, fb0, a00);
    a01 = wmma_f16(fa0, fb1, a01);
    a10 = wmma_f16(fa1, fb0, a10);
    a11 = wmma_f16(fa1, fb1, a11);
  }

  const int col = lane & 15, hi = (lane >> 4) & 1;
#pragma unroll
  for (int ti = 0; ti < 2; ++ti) {
#pragma unroll
    for (int tj = 0; tj < 2; ++tj) {
      const v8f acc = (ti == 0) ? ((tj == 0) ? a00 : a01)
                                : ((tj == 0) ? a10 : a11);
      const int c = n0 + tj * 16 + col;
      const float bias = bo[c];
#pragma unroll
      for (int r = 0; r < 8; ++r) {
        const int row = m0 + ti * 16 + r + 8 * hi;
        out[(size_t)row * CH + c] = acc[r] + bias + hs[(size_t)row * CH + c];
      }
    }
  }
}

// ---------------------------------------------------------------------------
// Host launcher
// ---------------------------------------------------------------------------
extern "C" void kernel_launch(void* const* d_in, const int* in_sizes, int n_in,
                              void* d_out, int out_size, void* d_ws, size_t ws_size,
                              hipStream_t stream) {
  (void)in_sizes; (void)n_in; (void)out_size; (void)ws_size;
  const float* hs = (const float*)d_in[0];
  const float* Wq = (const float*)d_in[1];
  const float* Wk = (const float*)d_in[2];
  const float* Wv = (const float*)d_in[3];
  const float* Wo = (const float*)d_in[4];
  const float* bo = (const float*)d_in[5];
  const long long* ids = (const long long*)d_in[6];
  float* out = (float*)d_out;

  char* ws = (char*)d_ws;
  size_t off = 0;
  auto carve = [&](size_t bytes) -> char* {
    char* p = ws + off;
    off += (bytes + 255) & ~(size_t)255;
    return p;
  };
  const size_t padBytes = (size_t)NB * NH * SEQ * DPAD * sizeof(_Float16); // 5,242,880
  _Float16* hs_h = (_Float16*)carve((size_t)ROWS * CH * sizeof(_Float16));
  _Float16* wtq  = (_Float16*)carve((size_t)CH * CH * sizeof(_Float16));
  _Float16* wtk  = (_Float16*)carve((size_t)CH * CH * sizeof(_Float16));
  _Float16* wtv  = (_Float16*)carve((size_t)CH * CH * sizeof(_Float16));
  _Float16* wto  = (_Float16*)carve((size_t)CH * CH * sizeof(_Float16));
  _Float16* Qp   = (_Float16*)carve(padBytes);
  _Float16* Kp   = (_Float16*)carve(padBytes);
  _Float16* Vt   = (_Float16*)carve(padBytes);
  _Float16* attn = (_Float16*)carve((size_t)ROWS * CH * sizeof(_Float16));

  // Zero padded Q/K/V (contiguous carve region, 256B-aligned sizes)
  const int n16 = (int)(3 * padBytes / 16);
  k_zero16<<<(n16 + 255) / 256, 256, 0, stream>>>((uint4*)Qp, n16);

  const int nconv = ROWS * CH;
  k_f32_to_f16<<<(nconv + 255) / 256, 256, 0, stream>>>(hs, hs_h, nconv);
  k_transpose_w<<<(4 * CH * CH + 255) / 256, 256, 0, stream>>>(
      Wq, Wk, Wv, Wo, wtq, wtk, wtv, wto);

  // 3 matrices * 160 mblocks * 10 nblocks = 4800 waves / 8 per block
  k_qkv_gemm<<<600, 256, 0, stream>>>(hs_h, wtq, wtk, wtv, Qp, Kp, Vt);

  // 5 * 8 * 64 q-tiles, one wave each
  k_attn<<<NB * NH * 64, 32, 0, stream>>>(Qp, Kp, Vt, ids, attn);

  // 160 mblocks * 10 nblocks = 1600 waves / 8 per block
  k_out_gemm<<<200, 256, 0, stream>>>(attn, wto, bo, hs, out);
}
